// MultiHeadAttn_28329604284592
// MI455X (gfx1250) — matrix operand order repaired and checked by reading, not compiled, not run on hardware
//
#include <hip/hip_runtime.h>
#include <hip/hip_bf16.h>

typedef __attribute__((ext_vector_type(16))) _Float16 v16h;
typedef __attribute__((ext_vector_type(8)))  _Float16 v8h;
typedef __attribute__((ext_vector_type(8)))  float    v8f;
typedef unsigned int u32x4 __attribute__((ext_vector_type(4)));
typedef int          i32x8 __attribute__((ext_vector_type(8)));
typedef int          i32x4 __attribute__((ext_vector_type(4)));

#define SEQ    1024
#define DIMC   768
#define NHEAD  12
#define DHE    64
#define NB     8

// ---------------------------------------------------------------------------
// WMMA fragment helpers (wave32; A/B use this file's contiguous k order, not the ISA table order)
//   A 16x32 : lane m=L%16, g=L/16 -> K run  [16g..16g+15]
//   B 32x16 : lane n=L%16, g=L/16 -> K run  [16g..16g+15]
//   C 16x16 : vgpr r, lane L      -> M = r + 8*(L/16), N = L%16
// ---------------------------------------------------------------------------

__device__ __forceinline__ v8f wmma_f16(v16h a, v16h b, v8f c) {
    return __builtin_amdgcn_wmma_f32_16x16x32_f16(
        false, a, false, b, (short)0, c, false, false);
}

__device__ __forceinline__ v16h combine16(v8h lo, v8h hi) {
    v16h a;
#pragma unroll
    for (int e = 0; e < 8; ++e) { a[e] = lo[e]; a[e + 8] = hi[e]; }
    return a;
}

__device__ __forceinline__ v16h load_frag_a_f16v(const _Float16* src, int ld, int k0, int lane) {
    const int m = lane & 15, g = (lane >> 4) & 1;
    const _Float16* p = src + (size_t)m * ld + k0 + 16 * g;
    return combine16(*(const v8h*)p, *(const v8h*)(p + 8));
}

__device__ __forceinline__ v16h load_frag_a_f32v(const float* src, int ld, int k0, int lane) {
    const int m = lane & 15, g = (lane >> 4) & 1;
    const float* p = src + (size_t)m * ld + k0 + 16 * g;
    const v8f lo = *(const v8f*)p;
    const v8f hi = *(const v8f*)(p + 8);
    v16h a;
#pragma unroll
    for (int e = 0; e < 8; ++e) { a[e] = (_Float16)lo[e]; a[e + 8] = (_Float16)hi[e]; }
    return a;
}

// B fragment where logical B[k][n] = src[n][k] (transposed source).
__device__ __forceinline__ v16h load_frag_b_trv(const _Float16* src, int ld, int k0, int n0, int lane) {
    const int nn = lane & 15, g = (lane >> 4) & 1;
    const _Float16* p = src + (size_t)(n0 + nn) * ld + k0 + 16 * g;
    return combine16(*(const v8h*)p, *(const v8h*)(p + 8));
}

// B fragment from pre-packed fragment-major weights: [frag][lane][16] halves.
__device__ __forceinline__ v16h load_frag_b_pk(const _Float16* packed, int frag, int lane) {
    const _Float16* p = packed + ((size_t)frag * 32 + lane) * 16;
    return combine16(*(const v8h*)p, *(const v8h*)(p + 8));
}

// ---------------------------------------------------------------------------
// Tensor Data Mover: 2D tile load Global -> LDS (CDNA5 ISA)
// D# group0: count=1 | lds_addr | global_addr[56:0] | type=2
// D# group1: data_size=2B, tensor_dim0/1, tile_dim0/1 (tile_dim2=0 -> 2D), stride0
// groups 2/3 zero (<=2D tensor). Tracked by TENSORcnt.
// ---------------------------------------------------------------------------
__device__ __forceinline__ unsigned lds_off32(const void* p) {
    // LDS aperture keeps the offset in addr[31:0] (ISA §10.2 aperture mapping)
    return (unsigned)(unsigned long long)p;
}

__device__ __forceinline__ void tdm_load_2d(unsigned lds_byte, const void* gptr,
                                            unsigned td0, unsigned td1,
                                            unsigned long long stride0_elems,
                                            unsigned tile0, unsigned tile1)
{
    const unsigned long long ga = (unsigned long long)gptr;
    u32x4 g0;
    g0[0] = 1u;                                                  // count=1, user D#
    g0[1] = lds_byte;                                            // lds_addr (bytes)
    g0[2] = (unsigned)ga;                                        // global_addr[31:0]
    g0[3] = (unsigned)((ga >> 32) & 0x1ffffffu) | (2u << 30);    // addr[56:32] | type=2
    i32x8 g1;
    g1[0] = (int)(1u << 16);                                     // data_size = 2 bytes
    g1[1] = (int)((td0 & 0xffffu) << 16);                        // tensor_dim0[15:0]
    g1[2] = (int)(((td0 >> 16) & 0xffffu) | ((td1 & 0xffffu) << 16));
    g1[3] = (int)(((td1 >> 16) & 0xffffu) | ((tile0 & 0xffffu) << 16));
    g1[4] = (int)(tile1 & 0xffffu);                              // tile_dim1, tile_dim2=0
    g1[5] = (int)(unsigned)(stride0_elems & 0xffffffffull);      // dim0 stride
    g1[6] = (int)(unsigned)((stride0_elems >> 32) & 0xffffull);
    g1[7] = 0;
    i32x4 z4 = {};
#if defined(__clang_major__) && __clang_major__ >= 23
    i32x8 z8 = {};
    __builtin_amdgcn_tensor_load_to_lds(g0, g1, z4, z4, z8, 0);
#else
    __builtin_amdgcn_tensor_load_to_lds(g0, g1, z4, z4, 0);
#endif
}

// ---------------------------------------------------------------------------
// Kernel 0: pack row-major f32 weight [K][N] into fragment-major f16.
// ---------------------------------------------------------------------------
__global__ void pack_kernel(const float* __restrict__ w, _Float16* __restrict__ out, int K, int N) {
    const int o = blockIdx.x * blockDim.x + threadIdx.x;
    if (o >= K * N) return;
    const int e      = o & 15;
    const int lane   = (o >> 4) & 31;
    const int rest   = o >> 9;
    const int ntiles = N >> 4;
    const int ntile  = rest % ntiles;
    const int kchunk = rest / ntiles;
    const int nn = lane & 15, g = (lane >> 4) & 1;
    const int kk = 16 * g + e;
    out[o] = (_Float16)w[(size_t)(kchunk * 32 + kk) * N + ntile * 16 + nn];
}

// ---------------------------------------------------------------------------
// Kernel 1: QKV projection. Q,K row-major [S][64] (LDS transpose, b128 stores);
// V stored transposed [64][S] (packed v8h stores).
// ---------------------------------------------------------------------------
__global__ __launch_bounds__(256) void qkv_kernel(
    const float* __restrict__ x,
    const _Float16* __restrict__ wqB, const _Float16* __restrict__ wkB, const _Float16* __restrict__ wvB,
    const float* __restrict__ bq, const float* __restrict__ bk, const float* __restrict__ bv,
    _Float16* __restrict__ Qb, _Float16* __restrict__ Kb, _Float16* __restrict__ VTb)
{
    __shared__ __attribute__((aligned(16))) _Float16 smem[8][16 * 64];
    const int lane = threadIdx.x & 31;
    const int wav  = threadIdx.x >> 5;
    const int wid  = blockIdx.x * 8 + wav;        // 0 .. 6143
    const int sb   = wid & 63;
    const int h    = (wid >> 6) % NHEAD;
    const int n    = wid / (64 * NHEAD);
    const int nl   = lane & 15;
    const int g    = (lane >> 4) & 1;
    const size_t head_off = ((size_t)(n * NHEAD + h)) * SEQ * DHE;

    const float* xb = x + ((size_t)(n * SEQ + sb * 16)) * DIMC + h * DHE;
    const v16h a0 = load_frag_a_f32v(xb, DIMC, 0,  lane);
    const v16h a1 = load_frag_a_f32v(xb, DIMC, 32, lane);

    _Float16* sp = smem[wav];

    const _Float16* WB[2] = {wqB, wkB};
    const float*    BI[2] = {bq, bk};
    _Float16*       DO[2] = {Qb, Kb};
#pragma unroll
    for (int w = 0; w < 2; ++w) {
#pragma unroll
        for (int t = 0; t < 4; ++t) {
            v8f c = {};
            c = wmma_f16(a0, load_frag_b_pk(WB[w], 0 + t, lane), c);
            c = wmma_f16(a1, load_frag_b_pk(WB[w], 4 + t, lane), c);
            const float bias = BI[w][t * 16 + nl];
#pragma unroll
            for (int r = 0; r < 8; ++r)
                sp[(8 * g + r) * 64 + t * 16 + nl] = (_Float16)(c[r] + bias);
        }
        __asm__ volatile("s_wait_dscnt 0" ::: "memory");
        _Float16* dst = DO[w] + head_off + (size_t)sb * 16 * DHE;
        const int row = lane >> 1, c0 = (lane & 1) * 32;
        const _Float16* s0 = sp + row * 64 + c0;
        _Float16* d0 = dst + row * 64 + c0;
#pragma unroll
        for (int i = 0; i < 4; ++i)
            *(v8h*)(d0 + i * 8) = *(const v8h*)(s0 + i * 8);
        __asm__ volatile("s_wait_dscnt 0" ::: "memory");
    }

#pragma unroll
    for (int t = 0; t < 4; ++t) {
        v8f c = {};
        c = wmma_f16(a0, load_frag_b_pk(wvB, 0 + t, lane), c);
        c = wmma_f16(a1, load_frag_b_pk(wvB, 4 + t, lane), c);
        const float bias = bv[t * 16 + nl];
        v8h pk8;
#pragma unroll
        for (int r = 0; r < 8; ++r) pk8[r] = (_Float16)(c[r] + bias);
        *(v8h*)(VTb + head_off + (size_t)(t * 16 + nl) * SEQ + sb * 16 + 8 * g) = pk8;
    }
}

// ---------------------------------------------------------------------------
// Kernel 2: flash attention + fused out-projection + channel permute.
// Block = 4 waves sharing one (n,h); K/V tiles staged in LDS by the TDM
// (tensor_load_to_lds), double-buffered, paced by TENSORcnt + barriers.
// ---------------------------------------------------------------------------
__global__ __launch_bounds__(128) void attn_kernel(
    const _Float16* __restrict__ Qb, const _Float16* __restrict__ Kb, const _Float16* __restrict__ VTb,
    const _Float16* __restrict__ wpB, const float* __restrict__ bp,
    float* __restrict__ out)
{
    __shared__ __attribute__((aligned(16))) _Float16 kvK[2][32 * 64];   // [buf][key][hidden]
    __shared__ __attribute__((aligned(16))) _Float16 kvV[2][64 * 32];   // [buf][col][key]
    __shared__ __attribute__((aligned(16))) _Float16 stage[4][16 * 64]; // per-wave P/O staging

    const int lane = threadIdx.x & 31;
    const int wav  = threadIdx.x >> 5;
    const int nh   = blockIdx.x >> 4;             // 0..95 -> (n,h)
    const int qg   = blockIdx.x & 15;
    const int qb   = qg * 4 + wav;                // 16-query block, 0..63
    const int h    = nh % NHEAD;
    const int n    = nh / NHEAD;
    const int nl   = lane & 15;
    const int g    = (lane >> 4) & 1;

    const size_t head_off = (size_t)nh * SEQ * DHE;
    const _Float16* Qh  = Qb  + head_off;
    const _Float16* Kh  = Kb  + head_off;
    const _Float16* VTh = VTb + head_off;
    const bool leader = (wav == 0);

    // prologue: DMA first K/V tiles while everyone loads Q fragments
    if (leader) {
        tdm_load_2d(lds_off32(kvK[0]), Kh,       DHE, 32, DHE, DHE, 32);
        tdm_load_2d(lds_off32(kvV[0]), VTh,      32, DHE, SEQ, 32, DHE);
    }
    const v16h aq0 = load_frag_a_f16v(Qh + (size_t)qb * 16 * DHE, DHE, 0,  lane);
    const v16h aq1 = load_frag_a_f16v(Qh + (size_t)qb * 16 * DHE, DHE, 32, lane);
    if (leader) __builtin_amdgcn_s_wait_tensorcnt(0);
    __syncthreads();

    float m_[8], l_[8];
    v8f o0 = {}, o1 = {}, o2 = {}, o3 = {};
#pragma unroll
    for (int r = 0; r < 8; ++r) { m_[r] = -1e30f; l_[r] = 0.0f; }

    _Float16* sp = stage[wav];
    int buf = 0;

    for (int kblk = 0; kblk < SEQ / 32; ++kblk) {
        if (leader && kblk + 1 < SEQ / 32) {
            tdm_load_2d(lds_off32(kvK[buf ^ 1]), Kh  + (size_t)(kblk + 1) * 32 * DHE,
                        DHE, 32, DHE, DHE, 32);
            tdm_load_2d(lds_off32(kvV[buf ^ 1]), VTh + (size_t)(kblk + 1) * 32,
                        32, DHE, SEQ, 32, DHE);
        }
        const _Float16* Kt = kvK[buf];
        const _Float16* Vt = kvV[buf];

        // scores S = Q(16x64) @ K^T(64x32), scale 1/sqrt(64); B frags from LDS
        v8f c0 = {}, c1 = {};
        c0 = wmma_f16(aq0, load_frag_b_trv(Kt, DHE, 0,  0,  lane), c0);
        c0 = wmma_f16(aq1, load_frag_b_trv(Kt, DHE, 32, 0,  lane), c0);
        c1 = wmma_f16(aq0, load_frag_b_trv(Kt, DHE, 0,  16, lane), c1);
        c1 = wmma_f16(aq1, load_frag_b_trv(Kt, DHE, 32, 16, lane), c1);

        float p0[8], p1[8], alpha[8];
#pragma unroll
        for (int r = 0; r < 8; ++r) {
            const float s0 = c0[r] * 0.125f;
            const float s1 = c1[r] * 0.125f;
            float mx = fmaxf(s0, s1);
            mx = fmaxf(mx, __shfl_xor(mx, 1, 32));
            mx = fmaxf(mx, __shfl_xor(mx, 2, 32));
            mx = fmaxf(mx, __shfl_xor(mx, 4, 32));
            mx = fmaxf(mx, __shfl_xor(mx, 8, 32));
            const float mn = fmaxf(m_[r], mx);
            alpha[r] = __expf(m_[r] - mn);
            p0[r] = __expf(s0 - mn);
            p1[r] = __expf(s1 - mn);
            float rs = p0[r] + p1[r];
            rs += __shfl_xor(rs, 1, 32);
            rs += __shfl_xor(rs, 2, 32);
            rs += __shfl_xor(rs, 4, 32);
            rs += __shfl_xor(rs, 8, 32);
            l_[r] = l_[r] * alpha[r] + rs;
            m_[r] = mn;
        }
#pragma unroll
        for (int r = 0; r < 8; ++r) {
            o0[r] *= alpha[r]; o1[r] *= alpha[r]; o2[r] *= alpha[r]; o3[r] *= alpha[r];
        }

        // P (C-layout f32) -> LDS -> A-layout f16
#pragma unroll
        for (int r = 0; r < 8; ++r) {
            const int M = 8 * g + r;
            sp[M * 32 + nl]      = (_Float16)p0[r];
            sp[M * 32 + 16 + nl] = (_Float16)p1[r];
        }
        __asm__ volatile("s_wait_dscnt 0" ::: "memory");
        const v16h ap = load_frag_a_f16v(sp, 32, 0, lane);

        // O += P(16x32) @ V(32x64); B[k][n] = Vt[n][k] from the LDS VT tile
        o0 = wmma_f16(ap, load_frag_b_trv(Vt, 32, 0, 0,  lane), o0);
        o1 = wmma_f16(ap, load_frag_b_trv(Vt, 32, 0, 16, lane), o1);
        o2 = wmma_f16(ap, load_frag_b_trv(Vt, 32, 0, 32, lane), o2);
        o3 = wmma_f16(ap, load_frag_b_trv(Vt, 32, 0, 48, lane), o3);

        if (leader && kblk + 1 < SEQ / 32) __builtin_amdgcn_s_wait_tensorcnt(0);
        __syncthreads();
        buf ^= 1;
    }

    // normalize + stage O (16x64) in LDS for the out-projection
#pragma unroll
    for (int r = 0; r < 8; ++r) {
        const int M = 8 * g + r;
        const float inv = 1.0f / l_[r];
        sp[M * 64 + 0  + nl] = (_Float16)(o0[r] * inv);
        sp[M * 64 + 16 + nl] = (_Float16)(o1[r] * inv);
        sp[M * 64 + 32 + nl] = (_Float16)(o2[r] * inv);
        sp[M * 64 + 48 + nl] = (_Float16)(o3[r] * inv);
    }
    __asm__ volatile("s_wait_dscnt 0" ::: "memory");
    const v16h ao0 = load_frag_a_f16v(sp, 64, 0,  lane);
    const v16h ao1 = load_frag_a_f16v(sp, 64, 32, lane);

    // out-proj Y = O @ wp + bp; write with channel permute c = e*NHEAD + h
#pragma unroll
    for (int t = 0; t < 4; ++t) {
        v8f cp = {};
        cp = wmma_f16(ao0, load_frag_b_pk(wpB, 0 + t, lane), cp);
        cp = wmma_f16(ao1, load_frag_b_pk(wpB, 4 + t, lane), cp);
        const int e = t * 16 + nl;
        const float bias = bp[e];
#pragma unroll
        for (int r = 0; r < 8; ++r) {
            const int M = 8 * g + r;
            const int srow = qb * 16 + M;
            out[((size_t)(n * SEQ + srow)) * DIMC + e * NHEAD + h] = cp[r] + bias;
        }
    }
}

// ---------------------------------------------------------------------------
// Kernel 3: LayerNorm over 768, f16 output. One wave per row.
// ---------------------------------------------------------------------------
__global__ __launch_bounds__(256) void ln_kernel(
    const float* __restrict__ xin, const float* __restrict__ gamma, const float* __restrict__ beta,
    _Float16* __restrict__ normed)
{
    const int lane = threadIdx.x & 31;
    const int wav  = threadIdx.x >> 5;
    const int row  = blockIdx.x * 8 + wav;
    const float* p = xin + (size_t)row * DIMC;

    float v[24];
    float s = 0.0f;
#pragma unroll
    for (int i = 0; i < 24; ++i) { v[i] = p[i * 32 + lane]; s += v[i]; }
#pragma unroll
    for (int mask = 16; mask > 0; mask >>= 1) s += __shfl_xor(s, mask, 32);
    const float mu = s * (1.0f / 768.0f);

    float var = 0.0f;
#pragma unroll
    for (int i = 0; i < 24; ++i) { const float d = v[i] - mu; var += d * d; }
#pragma unroll
    for (int mask = 16; mask > 0; mask >>= 1) var += __shfl_xor(var, mask, 32);
    const float inv = rsqrtf(var * (1.0f / 768.0f) + 1e-5f);

    _Float16* q = normed + (size_t)row * DIMC;
#pragma unroll
    for (int i = 0; i < 24; ++i) {
        const int c = i * 32 + lane;
        q[c] = (_Float16)((v[i] - mu) * inv * gamma[c] + beta[c]);
    }
}

// ---------------------------------------------------------------------------
// Kernel 4: y = residual + normed @ wff + bff (in-place on d_out, row-local)
// ---------------------------------------------------------------------------
__global__ __launch_bounds__(256) void ff_kernel(
    const _Float16* __restrict__ normed, const _Float16* __restrict__ wffB,
    const float* __restrict__ bff, float* __restrict__ y)
{
    const int lane = threadIdx.x & 31;
    const int wav  = threadIdx.x >> 5;
    const int wid  = blockIdx.x * 8 + wav;
    const int ng   = wid % 12;
    const int mb   = wid / 12;
    const int nl   = lane & 15;
    const int g    = (lane >> 4) & 1;
    const int n0   = ng * 64;

    const _Float16* arow = normed + (size_t)mb * 16 * DIMC;
    v8f c0 = {}, c1 = {}, c2 = {}, c3 = {};

    for (int kc = 0; kc < 24; ++kc) {
        if (kc + 1 < 24) {
            const _Float16* pf = wffB + ((size_t)((kc + 1) * 48 + ng * 4) * 32) * 16 + lane * 64;
            __builtin_prefetch((const void*)pf, 0, 1);
        }
        const v16h a = load_frag_a_f16v(arow, DIMC, kc * 32, lane);
        const int fbase = kc * 48 + ng * 4;
        c0 = wmma_f16(a, load_frag_b_pk(wffB, fbase + 0, lane), c0);
        c1 = wmma_f16(a, load_frag_b_pk(wffB, fbase + 1, lane), c1);
        c2 = wmma_f16(a, load_frag_b_pk(wffB, fbase + 2, lane), c2);
        c3 = wmma_f16(a, load_frag_b_pk(wffB, fbase + 3, lane), c3);
    }

#pragma unroll
    for (int t = 0; t < 4; ++t) {
        const v8f* ct = (t == 0) ? &c0 : (t == 1) ? &c1 : (t == 2) ? &c2 : &c3;
        const float bias = bff[n0 + t * 16 + nl];
#pragma unroll
        for (int r = 0; r < 8; ++r) {
            const int M = 8 * g + r;
            const size_t idx = ((size_t)(mb * 16 + M)) * DIMC + n0 + t * 16 + nl;
            y[idx] = y[idx] + (*ct)[r] + bias;
        }
    }
}

// ---------------------------------------------------------------------------
// Host launcher
// ---------------------------------------------------------------------------
extern "C" void kernel_launch(void* const* d_in, const int* in_sizes, int n_in,
                              void* d_out, int out_size, void* d_ws, size_t ws_size,
                              hipStream_t stream) {
    (void)in_sizes; (void)n_in; (void)out_size; (void)ws_size;
    const float* x     = (const float*)d_in[0];
    const float* wq    = (const float*)d_in[1];
    const float* bq    = (const float*)d_in[2];
    const float* wk    = (const float*)d_in[3];
    const float* bk    = (const float*)d_in[4];
    const float* wv    = (const float*)d_in[5];
    const float* bv    = (const float*)d_in[6];
    const float* wp    = (const float*)d_in[7];
    const float* bp    = (const float*)d_in[8];
    const float* gamma = (const float*)d_in[9];
    const float* beta  = (const float*)d_in[10];
    const float* wff   = (const float*)d_in[11];
    const float* bff   = (const float*)d_in[12];
    float* y = (float*)d_out;

    char* ws = (char*)d_ws;
    _Float16* wffB = (_Float16*)ws;              ws += (size_t)DIMC * DIMC * 2;
    _Float16* wqB  = (_Float16*)ws;              ws += (size_t)DHE * DHE * 2;
    _Float16* wkB  = (_Float16*)ws;              ws += (size_t)DHE * DHE * 2;
    _Float16* wvB  = (_Float16*)ws;              ws += (size_t)DHE * DHE * 2;
    _Float16* wpB  = (_Float16*)ws;              ws += (size_t)DHE * DHE * 2;
    const size_t qkv_elems = (size_t)NB * NHEAD * SEQ * DHE;
    _Float16* Qb  = (_Float16*)ws;               ws += qkv_elems * 2;
    _Float16* Kb  = (_Float16*)ws;               ws += qkv_elems * 2;
    _Float16* VTb = (_Float16*)ws;               ws += qkv_elems * 2;
    _Float16* normed = (_Float16*)ws;            ws += (size_t)NB * SEQ * DIMC * 2;

    pack_kernel<<<(DIMC * DIMC + 255) / 256, 256, 0, stream>>>(wff, wffB, DIMC, DIMC);
    pack_kernel<<<16, 256, 0, stream>>>(wq, wqB, DHE, DHE);
    pack_kernel<<<16, 256, 0, stream>>>(wk, wkB, DHE, DHE);
    pack_kernel<<<16, 256, 0, stream>>>(wv, wvB, DHE, DHE);
    pack_kernel<<<16, 256, 0, stream>>>(wp, wpB, DHE, DHE);

    qkv_kernel<<<768, 256, 0, stream>>>(x, wqB, wkB, wvB, bq, bk, bv, Qb, Kb, VTb);
    attn_kernel<<<1536, 128, 0, stream>>>(Qb, Kb, VTb, wpB, bp, y);
    ln_kernel<<<1024, 256, 0, stream>>>(y, gamma, beta, normed);
    ff_kernel<<<768, 256, 0, stream>>>(normed, wffB, bff, y);
}
